// RGCN_78847009620170
// MI455X (gfx1250) — compile-verified, matrix operand order checked by reading, not run on hardware
//
#include <hip/hip_runtime.h>
#include <hip/hip_bf16.h>

#define N_NODES 50000
#define N_EDGES 500000
#define DIM 128
#define N_REL 51
#define N_LAYERS 4
#define BQ 1024
#define KQ 32
#define KTOT (13 * DIM)    // 1664
#define KSTEPS (KTOT / 32) // 52

typedef __attribute__((ext_vector_type(16))) _Float16 v16h;
typedef __attribute__((ext_vector_type(8)))  float    v8f;

// ---- monotone float <-> uint mapping so we can use hardware integer atomics for max/min ----
__device__ __forceinline__ unsigned ford(float f) {
  unsigned u = __float_as_uint(f);
  return (u & 0x80000000u) ? ~u : (u | 0x80000000u);
}
__device__ __forceinline__ float funord(unsigned u) {
  return __uint_as_float((u & 0x80000000u) ? (u & 0x7fffffffu) : ~u);
}

// WMMA 16-bit operand register layout (CDNA5 ISA 7.12.2):
// lane = (row_or_col & 15) + 16*hi, half h, with kq = 16*(h>>3) + 8*hi + (h&7)
__device__ __forceinline__ int kq_of(int h, int hi) {
  return ((h & 8) << 1) + (hi << 3) + (h & 7);
}

// ------------------------------------------------------------------ degree / scales
__global__ void k_init_deg(unsigned* deg, float* scaleSum) {
  int i = blockIdx.x * blockDim.x + threadIdx.x;
  if (i < N_NODES) deg[i] = 0u;
  if (i == 0) *scaleSum = 0.f;
}

__global__ void k_count_deg(const int* __restrict__ ed, unsigned* deg) {
  int i = blockIdx.x * blockDim.x + threadIdx.x;
  if (i < N_EDGES) atomicAdd(&deg[ed[i]], 1u);
}

__global__ void k_scale1(const unsigned* __restrict__ deg, float* degf,
                         float* scaleTmp, float* scaleSum) {
  int i = blockIdx.x * blockDim.x + threadIdx.x;
  if (i < N_NODES) {
    float d = (float)deg[i] + 1.0f;
    degf[i] = d;
    float s = logf(d);
    scaleTmp[i] = s;
    atomicAdd(scaleSum, s);
  }
}

__global__ void k_scale2(const float* __restrict__ scaleTmp,
                         const float* __restrict__ scaleSum, float* scales) {
  int i = blockIdx.x * blockDim.x + threadIdx.x;
  if (i < N_NODES) {
    float mean = *scaleSum * (1.0f / (float)N_NODES);
    float sc = scaleTmp[i] / mean;
    scales[3 * i + 0] = 1.0f;
    scales[3 * i + 1] = sc;
    scales[3 * i + 2] = 1.0f / fmaxf(sc, 1e-2f);
  }
}

// ------------------------------------------------------------------ per-layer accumulator init (boundary = x0)
__global__ void k_acc_init(const float* __restrict__ x0, float* accSum, float* accSq,
                           unsigned* accMax, unsigned* accMin) {
  int i = blockIdx.x * blockDim.x + threadIdx.x;
  if (i < N_NODES * DIM) {
    float v = x0[i];
    accSum[i] = v;
    accSq[i]  = v * v;
    unsigned k = ford(v);
    accMax[i] = k;
    accMin[i] = k;
  }
}

// ------------------------------------------------------------------ lin_w fp32 -> fp16 swizzled into WMMA B-fragment order
// out[((kb*8 + wave)*32 + lane)*16 + h] = lin_w[(kb*32 + kq(h,lane>>4)) * 128 + wave*16 + (lane&15)]
__global__ void k_w16s(const float* __restrict__ w, _Float16* __restrict__ w16s) {
  int o = blockIdx.x * blockDim.x + threadIdx.x;
  if (o < KTOT * DIM) {
    const int h    = o & 15;
    const int lane = (o >> 4) & 31;
    const int t2   = o >> 9;
    const int wv   = t2 & 7;
    const int kb   = t2 >> 3;
    const int kq   = kq_of(h, lane >> 4);
    const int col  = wv * 16 + (lane & 15);
    w16s[o] = (_Float16)w[(size_t)(kb * 32 + kq) * DIM + col];
  }
}

// ------------------------------------------------------------------ edge message + scatter (wave per edge)
__global__ void k_edge(const float* __restrict__ x, const float* __restrict__ relw,
                       const int* __restrict__ es, const int* __restrict__ ed,
                       const int* __restrict__ et, const float* __restrict__ ew,
                       float* accSum, float* accSq, unsigned* accMax, unsigned* accMin) {
  const int lane = threadIdx.x & 31;
  const int gw = (blockIdx.x * blockDim.x + threadIdx.x) >> 5;
  const int nw = (gridDim.x * blockDim.x) >> 5;
  for (int e = gw; e < N_EDGES; e += nw) {
    const int si = es[e], di = ed[e], rt = et[e];
    const float w = ew[e];
    const float4 xv = *(const float4*)(x    + (size_t)si * DIM + lane * 4);
    const float4 rv = *(const float4*)(relw + (size_t)rt * DIM + lane * 4);
    float mv[4] = { xv.x * rv.x * w, xv.y * rv.y * w, xv.z * rv.z * w, xv.w * rv.w * w };
    const int base = di * DIM + lane * 4;
#pragma unroll
    for (int q = 0; q < 4; ++q) {
      atomicAdd(&accSum[base + q], mv[q]);
      atomicAdd(&accSq[base + q],  mv[q] * mv[q]);
      atomicMax(&accMax[base + q], ford(mv[q]));
      atomicMin(&accMin[base + q], ford(mv[q]));
    }
  }
}

// ------------------------------------------------------------------ fused feature-build + WMMA GEMM + LN/ReLU/residual
// Block: 256 threads = 8 wave32, 16 nodes x 128 outputs. K = 1664 in 52 steps of 32.
// A tile is built once, pre-scaled, directly in WMMA fragment order in LDS.
// B fragments stream from the pre-swizzled fp16 weight panel (L2-resident, 426KB).
__launch_bounds__(256, 2)
__global__ void k_gemm(const float* __restrict__ accSum, const float* __restrict__ accSq,
                       const unsigned* __restrict__ accMax, const unsigned* __restrict__ accMin,
                       const float* __restrict__ degf, const float* __restrict__ scales,
                       const float* __restrict__ xcur, const _Float16* __restrict__ w16s,
                       const float* __restrict__ bias, const float* __restrict__ gamma,
                       const float* __restrict__ beta, float* __restrict__ xnext) {
  __shared__ _Float16 Aswz[KSTEPS * 32 * 16];  // 52KB: A fragments in register layout
  __shared__ float    Ds[16 * 132];            // output tile for LayerNorm stage

  const int t = threadIdx.x;
  const int node0 = blockIdx.x * 16;

  // ---- stage 1: features -> pre-scaled fp16 A tile, written directly in fragment order ----
  for (int idx = t; idx < 16 * DIM; idx += 256) {
    const int n = idx >> 7, d = idx & 127;
    const int g = (node0 + n) * DIM + d;
    const float dg   = degf[node0 + n];
    const float mean = accSum[g] / dg;
    const float sqm  = accSq[g] / dg;
    const float sd   = sqrtf(fmaxf(sqm - mean * mean, 1e-6f));
    const float f4[4] = { mean, funord(accMax[g]), funord(accMin[g]), sd };
    const float sc[3] = { scales[(node0 + n) * 3 + 0],
                          scales[(node0 + n) * 3 + 1],
                          scales[(node0 + n) * 3 + 2] };
    const float xv = xcur[g];
#pragma unroll
    for (int j = 0; j < 13; ++j) {
      const float av = (j < 12) ? f4[j & 3] * sc[j >> 2] : xv;
      const int K  = j * DIM + d;
      const int kb = K >> 5, kq = K & 31;
      const int hi = (kq >> 3) & 1;
      const int h  = ((kq >> 4) << 3) + (kq & 7);
      const int lane = n + (hi << 4);
      Aswz[(kb * 32 + lane) * 16 + h] = (_Float16)av;
    }
  }
  __syncthreads();

  const int lane = t & 31;
  const int wave = t >> 5;          // 0..7 -> 16-column tile
  const int mrow = lane & 15;
  const int hi   = lane >> 4;
  const int ncol = wave * 16 + mrow;

  // ---- stage 2: barrier-free, branch-free WMMA K-loop ----
  // per step: 2x ds_load_b128 (A) + 2x global_load_b128 (B, L2-resident) + 1 wmma
  const _Float16* aptr = &Aswz[lane * 16];
  const _Float16* bptr = w16s + ((size_t)wave * 32 + lane) * 16;
  v8f acc = {};
#pragma unroll 4
  for (int kb = 0; kb < KSTEPS; ++kb) {
    v16h a = *(const v16h*)(aptr + kb * (32 * 16));
    v16h b = *(const v16h*)(bptr + (size_t)kb * (8 * 32 * 16));
    acc = __builtin_amdgcn_wmma_f32_16x16x32_f16(false, a, false, b, (short)0, acc, false, false);
  }

  // ---- stage 3: spill D tile + bias ----
#pragma unroll
  for (int r = 0; r < 8; ++r) {
    const int m = r + (hi << 3);
    Ds[m * 132 + ncol] = acc[r] + bias[ncol];
  }
  __syncthreads();

  // ---- stage 4: LayerNorm + ReLU + residual (16 lanes per row, shfl_xor reduce) ----
  {
    const int row = t >> 4;
    const int sub = t & 15;
    float v[8], s0 = 0.f, s1 = 0.f;
#pragma unroll
    for (int q = 0; q < 8; ++q) {
      v[q] = Ds[row * 132 + sub * 8 + q];
      s0 += v[q]; s1 += v[q] * v[q];
    }
#pragma unroll
    for (int msk = 1; msk < 16; msk <<= 1) {
      s0 += __shfl_xor(s0, msk, 32);
      s1 += __shfl_xor(s1, msk, 32);
    }
    const float mu   = s0 * (1.f / 128.f);
    const float var  = s1 * (1.f / 128.f) - mu * mu;
    const float rstd = rsqrtf(var + 1e-5f);
    const int node = node0 + row;
#pragma unroll
    for (int q = 0; q < 8; ++q) {
      const int c = sub * 8 + q;
      float o = (v[q] - mu) * rstd * gamma[c] + beta[c];
      o = fmaxf(o, 0.f);
      o += xcur[node * DIM + c];
      xnext[node * DIM + c] = o;
    }
  }
}

// ------------------------------------------------------------------ DistMult scoring (wave per triple)
__global__ void k_score(const float* __restrict__ x, const float* __restrict__ qw,
                        const int* __restrict__ src, const int* __restrict__ rel,
                        const int* __restrict__ dst, float* __restrict__ out) {
  const int lane = threadIdx.x & 31;
  const int gw = (blockIdx.x * blockDim.x + threadIdx.x) >> 5;
  if (gw >= BQ * KQ) return;
  const int si = src[gw], ri = rel[gw], di = dst[gw];
  const float4 a = *(const float4*)(x  + (size_t)si * DIM + lane * 4);
  const float4 q = *(const float4*)(qw + (size_t)ri * DIM + lane * 4);
  const float4 b = *(const float4*)(x  + (size_t)di * DIM + lane * 4);
  float s = a.x * q.x * b.x + a.y * q.y * b.y + a.z * q.z * b.z + a.w * q.w * b.w;
#pragma unroll
  for (int m = 1; m < 32; m <<= 1) s += __shfl_xor(s, m, 32);
  if (lane == 0) out[gw] = s;
}

// ------------------------------------------------------------------ host launcher
extern "C" void kernel_launch(void* const* d_in, const int* in_sizes, int n_in,
                              void* d_out, int out_size, void* d_ws, size_t ws_size,
                              hipStream_t stream) {
  const float* x0      = (const float*)d_in[0];
  const int*   ei      = (const int*)  d_in[1];
  const int*   etype   = (const int*)  d_in[2];
  const float* ew      = (const float*)d_in[3];
  const float* rel_w   = (const float*)d_in[4];
  const float* lin_w   = (const float*)d_in[5];
  const float* lin_b   = (const float*)d_in[6];
  const float* ln_g    = (const float*)d_in[7];
  const float* ln_b    = (const float*)d_in[8];
  const float* query_w = (const float*)d_in[9];
  const int*   srcq    = (const int*)  d_in[10];
  const int*   relq    = (const int*)  d_in[11];
  const int*   dstq    = (const int*)  d_in[12];
  float* out = (float*)d_out;

  // workspace carve-out (256B aligned regions)
  char* p = (char*)d_ws;
  auto alloc = [&](size_t bytes) -> char* {
    char* r = p;
    p += (bytes + 255) & ~(size_t)255;
    return r;
  };
  unsigned* deg      = (unsigned*)alloc((size_t)N_NODES * 4);
  float*    degf     = (float*)   alloc((size_t)N_NODES * 4);
  float*    scaleTmp = (float*)   alloc((size_t)N_NODES * 4);
  float*    scaleSum = (float*)   alloc(256);
  float*    scales   = (float*)   alloc((size_t)N_NODES * 3 * 4);
  float*    accSum   = (float*)   alloc((size_t)N_NODES * DIM * 4);
  float*    accSq    = (float*)   alloc((size_t)N_NODES * DIM * 4);
  unsigned* accMax   = (unsigned*)alloc((size_t)N_NODES * DIM * 4);
  unsigned* accMin   = (unsigned*)alloc((size_t)N_NODES * DIM * 4);
  float*    xA       = (float*)   alloc((size_t)N_NODES * DIM * 4);
  float*    xB       = (float*)   alloc((size_t)N_NODES * DIM * 4);
  _Float16* w16s     = (_Float16*)alloc((size_t)KTOT * DIM * 2);

  const int* es = ei;             // edge_index[0] = source
  const int* ed = ei + N_EDGES;   // edge_index[1] = destination

  k_init_deg<<<(N_NODES + 255) / 256, 256, 0, stream>>>(deg, scaleSum);
  k_count_deg<<<(N_EDGES + 255) / 256, 256, 0, stream>>>(ed, deg);
  k_scale1<<<(N_NODES + 255) / 256, 256, 0, stream>>>(deg, degf, scaleTmp, scaleSum);
  k_scale2<<<(N_NODES + 255) / 256, 256, 0, stream>>>(scaleTmp, scaleSum, scales);

  hipMemcpyAsync(xA, x0, (size_t)N_NODES * DIM * 4, hipMemcpyDeviceToDevice, stream);

  float* xcur = xA;
  float* xnext = xB;
  for (int l = 0; l < N_LAYERS; ++l) {
    k_acc_init<<<(N_NODES * DIM + 255) / 256, 256, 0, stream>>>(x0, accSum, accSq, accMax, accMin);
    k_w16s<<<(KTOT * DIM + 255) / 256, 256, 0, stream>>>(lin_w + (size_t)l * KTOT * DIM, w16s);
    k_edge<<<2048, 256, 0, stream>>>(xcur, rel_w + (size_t)l * N_REL * DIM,
                                     es, ed, etype, ew, accSum, accSq, accMax, accMin);
    k_gemm<<<N_NODES / 16, 256, 0, stream>>>(accSum, accSq, accMax, accMin, degf, scales,
                                             xcur, w16s,
                                             lin_b + (size_t)l * DIM,
                                             ln_g + (size_t)l * DIM,
                                             ln_b + (size_t)l * DIM, xnext);
    float* t2 = xcur; xcur = xnext; xnext = t2;
  }

  k_score<<<(BQ * KQ * 32) / 256, 256, 0, stream>>>(xcur, query_w, srcq, relq, dstq, out);
}
